// Attn_52828097741235
// MI455X (gfx1250) — matrix operand order repaired and checked by reading, not compiled, not run on hardware
//
#include <hip/hip_runtime.h>
#include <hip/hip_bf16.h>

// Problem constants (reference: T=4096, B=16, H=512)
#define KT 4096
#define KB 16
#define KH 512
#define SKP (KH + 8)   // padded LDS row stride in bf16 elems: 1040B = 260 dwords -> 4-bank skew/row

typedef __attribute__((ext_vector_type(16))) __bf16 v16bf;
typedef __attribute__((ext_vector_type(8)))  float  v8f;

struct U32x8 { uint4 lo; uint4 hi; };   // 32B carrier for a v16bf fragment

__device__ __forceinline__ unsigned short f2bf(float x) {
  unsigned u = __float_as_uint(x);
  u += 0x7FFFu + ((u >> 16) & 1u);          // round-to-nearest-even
  return (unsigned short)(u >> 16);
}
__device__ __forceinline__ unsigned pack2bf(float a, float b) {
  return (unsigned)f2bf(a) | ((unsigned)f2bf(b) << 16);
}

// Branch-free tanh. Prefer gfx1250's hardware V_TANH_F32 if the builtin
// exists in this toolchain; otherwise 4 cheap ops on raw trans instructions:
//   t = exp2(-2*log2(e)*|x|);  tanh(x) = sign(x) * (1-t)/(1+t)
// No EXEC divergence (the OCML tanhf path costs ~45 VALU ops + saveexec).
__device__ __forceinline__ float fast_tanh(float x) {
#if __has_builtin(__builtin_amdgcn_tanhf)
  return __builtin_amdgcn_tanhf(x);
#elif __has_builtin(__builtin_amdgcn_tanh_f32)
  return __builtin_amdgcn_tanh_f32(x);
#else
  float ax = __builtin_fabsf(x);
  float t  = __builtin_amdgcn_exp2f(ax * -2.8853900817779268f); // -2/ln(2)^-1 = -2*log2(e)
  float r  = (1.0f - t) * __builtin_amdgcn_rcpf(1.0f + t);
  return __builtin_copysignf(r, x);
#endif
}

// ---------------------------------------------------------------------------
// Pass 0a: g[b][h] = hidden[b] @ Wh.T + bias   (Wh[h,k] = W[k,h])
__global__ __launch_bounds__(256) void compute_g(
    const float* __restrict__ hidden, const float* __restrict__ W,
    const float* __restrict__ bias, float* __restrict__ g)
{
  int idx = blockIdx.x * 256 + threadIdx.x;     // 0 .. KB*KH-1
  int b = idx >> 9, h = idx & (KH - 1);
  const float* hr = hidden + (size_t)b * KH;
  float s = bias[h];
  for (int k = 0; k < KH; ++k)
    s += hr[k] * W[(size_t)k * (2 * KH) + h];   // coalesced across h
  g[idx] = s;
}

// ---------------------------------------------------------------------------
// Pass 0b: pack We (= W[:, H:]) into bf16 WMMA-B fragment layout.
// frag = n_tile*16 + ks.  Per ISA 7.12.2 (16-bit B, 32x16):
//   lanes 0-15:  N = n_tile*16+lane, halves 0..7 = K ks*32+0..7,  8..15 = K ks*32+16..23
//   lanes 16-31: same N,              halves 0..7 = K ks*32+8..15, 8..15 = K ks*32+24..31
__global__ __launch_bounds__(256) void pack_we(
    const float* __restrict__ W, unsigned short* __restrict__ wfrag)
{
  int idx  = blockIdx.x * 256 + threadIdx.x;    // 32 ntiles *16 ks *32 lanes = 16384
  int lane = idx & 31;
  int frag = idx >> 5;
  int ks = frag & 15, n_tile = frag >> 4;
  int n = n_tile * 16 + (lane & 15);
  int kbase = ks * 32 + ((lane < 16) ? 0 : 8);
  unsigned short* dst = wfrag + ((size_t)frag * 32 + lane) * 16;
  #pragma unroll
  for (int j = 0; j < 16; ++j)
    dst[j] = f2bf(W[(size_t)(kbase + (j < 8 ? j : j + 8)) * (2 * KH) + KH + n]);
}

// ---------------------------------------------------------------------------
// Pass 1: scores[b][t] = ( tanh(enc_b @ We.T + g_b) ) @ v   via bf16 WMMA.
// Block: 256 thr = 8 waves. Tile M=64 (t rows, one batch), N=512, K=512.
// Waves: (wave>>1) -> M-group of 16 rows, (wave&1) -> N half (16 n-tiles).
__global__ __launch_bounds__(256) void attn_energy(
    const float* __restrict__ enc,                 // [T][B][H]
    const unsigned short* __restrict__ wfrag,      // packed B frags
    const float* __restrict__ g,                   // [B][H]
    const float* __restrict__ v,                   // [H]
    float* __restrict__ scores)                    // [B][T]
{
  const int tb = blockIdx.x, b = blockIdx.y;
  const int tid = threadIdx.x;
  const int wave = tid >> 5, lane = tid & 31;
  const int t0 = tb * 64;

  extern __shared__ unsigned char smem[];
  unsigned short* As = (unsigned short*)smem;                  // [64][SKP] bf16
  float* g_s = (float*)(smem + (size_t)64 * SKP * 2);          // [512]
  float* v_s = g_s + KH;                                       // [512]
  float* sc  = v_s + KH;                                       // [2][64]

  for (int i = tid; i < KH; i += 256) { g_s[i] = g[(size_t)b * KH + i]; v_s[i] = v[i]; }

  // Stage A tile: 64 rows of enc[t][b][:], f32 -> bf16 (padded rows).
  for (int i = tid; i < 64 * (KH / 4); i += 256) {
    int r = i >> 7, c4 = i & 127;                              // KH/4 = 128
    float4 f = ((const float4*)(enc + (size_t)(t0 + r) * (KB * KH) + (size_t)b * KH))[c4];
    uint2 p; p.x = pack2bf(f.x, f.y); p.y = pack2bf(f.z, f.w);
    *(uint2*)(As + (size_t)r * SKP + c4 * 4) = p;              // 8B aligned
  }
  __syncthreads();

  const int m_base = (wave >> 1) * 16;
  const int n_half = wave & 1;
  const int lm  = lane & 15;
  const int sel = (lane < 16) ? 0 : 8;                         // ISA 16-bit A layout
  const unsigned short* arow = As + (size_t)(m_base + lm) * SKP;

  float rowsum[8];
  #pragma unroll
  for (int r = 0; r < 8; ++r) rowsum[r] = 0.f;

  for (int nt = 0; nt < 16; ++nt) {
    const int n_tile = n_half * 16 + nt;
    const unsigned short* bbase =
        wfrag + (size_t)(n_tile * 16) * 512 + (size_t)lane * 16;
    v8f acc = {};
    #pragma unroll 4
    for (int ks = 0; ks < 16; ++ks) {
      const int k0 = ks * 32;
      U32x8 araw;
      araw.lo = *(const uint4*)(arow + k0 + sel);              // K = k0+sel .. +7
      araw.hi = *(const uint4*)(arow + k0 + 16 + sel);         // K = k0+16+sel .. +7
      v16bf a  = __builtin_bit_cast(v16bf, araw);
      v16bf bm = *(const v16bf*)(bbase + (size_t)ks * 512);    // contiguous 32B/lane
      acc = __builtin_amdgcn_wmma_f32_16x16x32_bf16(
          false, a, false, bm, (short)0, acc, false, false);
    }
    // Fused epilogue: C[m][n] layout -> VGPR r: M = r + 8*(lane>=16), N = lm
    const int nc = n_tile * 16 + lm;
    const float gb = g_s[nc], vv = v_s[nc];
    #pragma unroll
    for (int r = 0; r < 8; ++r)
      rowsum[r] += fast_tanh(acc[r] + gb) * vv;
  }

  // Reduce each row-sum over the 16 lanes holding that row's columns.
  #pragma unroll
  for (int r = 0; r < 8; ++r) {
    float s = rowsum[r];
    s += __shfl_xor(s, 1, 32);
    s += __shfl_xor(s, 2, 32);
    s += __shfl_xor(s, 4, 32);
    s += __shfl_xor(s, 8, 32);
    rowsum[r] = s;
  }
  if (lm == 0) {
    int rb = m_base + ((lane >= 16) ? 8 : 0);
    #pragma unroll
    for (int r = 0; r < 8; ++r) sc[n_half * 64 + rb + r] = rowsum[r]; // unique slots
  }
  __syncthreads();
  if (tid < 64)
    scores[(size_t)b * KT + t0 + tid] = sc[tid] + sc[64 + tid];
}

// ---------------------------------------------------------------------------
// Pass 2: softmax over T per batch.
__global__ __launch_bounds__(256) void softmax_t(
    const float* __restrict__ scores, float* __restrict__ attn)
{
  const int b = blockIdx.x, tid = threadIdx.x;
  __shared__ float red[256];
  const float* s = scores + (size_t)b * KT;
  float m = -3.402823466e38f;
  for (int t = tid; t < KT; t += 256) m = fmaxf(m, s[t]);
  red[tid] = m; __syncthreads();
  for (int st = 128; st; st >>= 1) { if (tid < st) red[tid] = fmaxf(red[tid], red[tid + st]); __syncthreads(); }
  m = red[0]; __syncthreads();
  float sum = 0.f;
  for (int t = tid; t < KT; t += 256) sum += __expf(s[t] - m);
  red[tid] = sum; __syncthreads();
  for (int st = 128; st; st >>= 1) { if (tid < st) red[tid] += red[tid + st]; __syncthreads(); }
  const float inv = 1.f / red[0];
  for (int t = tid; t < KT; t += 256)
    attn[(size_t)b * KT + t] = __expf(s[t] - m) * inv;
}

// ---------------------------------------------------------------------------
// Pass 3: partial ctx over T-chunks (keeps >16 WGPs busy).
__global__ __launch_bounds__(256) void ctx_partial(
    const float* __restrict__ enc, const float* __restrict__ attn,
    float* __restrict__ ctxp)                     // [B][16][H]
{
  const int chunk = blockIdx.x, b = blockIdx.y, tid = threadIdx.x;
  float a0 = 0.f, a1 = 0.f;
  const int ts = chunk * (KT / 16);
  for (int t = ts; t < ts + KT / 16; ++t) {
    const float w = attn[(size_t)b * KT + t];     // uniform -> scalar load
    const float* row = enc + (size_t)t * (KB * KH) + (size_t)b * KH;
    a0 += w * row[tid];
    a1 += w * row[tid + 256];
  }
  float* dst = ctxp + ((size_t)b * 16 + chunk) * KH;
  dst[tid] = a0; dst[tid + 256] = a1;
}

__global__ __launch_bounds__(256) void ctx_final(
    const float* __restrict__ ctxp, float* __restrict__ out)
{
  const int idx = blockIdx.x * 256 + threadIdx.x; // B*H = 8192
  const int b = idx >> 9, h = idx & (KH - 1);
  float s = 0.f;
  #pragma unroll
  for (int c = 0; c < 16; ++c) s += ctxp[((size_t)b * 16 + c) * KH + h];
  out[idx] = s;
}

// ---------------------------------------------------------------------------
extern "C" void kernel_launch(void* const* d_in, const int* in_sizes, int n_in,
                              void* d_out, int out_size, void* d_ws, size_t ws_size,
                              hipStream_t stream)
{
  (void)in_sizes; (void)n_in; (void)out_size; (void)ws_size;
  const float* hidden = (const float*)d_in[0];   // (1,B,H)
  const float* enc    = (const float*)d_in[1];   // (T,B,H)
  const float* W      = (const float*)d_in[2];   // (H,2H)
  const float* bias   = (const float*)d_in[3];   // (H,)
  const float* v      = (const float*)d_in[4];   // (H,)
  float* out = (float*)d_out;                    // (1,B,H)

  char* ws = (char*)d_ws;
  float*          g      = (float*)ws;          ws += (size_t)KB * KH * 4;          //  32 KB
  unsigned short* wfrag  = (unsigned short*)ws; ws += (size_t)KH * KH * 2;          // 512 KB
  float*          scores = (float*)ws;          ws += (size_t)KB * KT * 4;          // 256 KB
  float*          attn   = (float*)ws;          ws += (size_t)KB * KT * 4;          // 256 KB
  float*          ctxp   = (float*)ws;                                              // 512 KB

  compute_g<<<dim3((KB * KH) / 256), 256, 0, stream>>>(hidden, W, bias, g);
  pack_we<<<dim3((32 * 16 * 32) / 256), 256, 0, stream>>>(W, wfrag);

  const size_t shmem = (size_t)64 * SKP * 2 + KH * 4 + KH * 4 + 2 * 64 * 4; // ~71 KB
  attn_energy<<<dim3(KT / 64, KB), 256, shmem, stream>>>(enc, wfrag, g, v, scores);

  softmax_t<<<dim3(KB), 256, 0, stream>>>(scores, attn);
  ctx_partial<<<dim3(16, KB), 256, 0, stream>>>(enc, attn, ctxp);
  ctx_final<<<dim3((KB * KH) / 256), 256, 0, stream>>>(ctxp, out);
}
